// AttnBlock_74663711474072
// MI455X (gfx1250) — matrix operand order repaired and checked by reading, not compiled, not run on hardware
//
#include <hip/hip_runtime.h>
#include <hip/hip_bf16.h>

#define BB  8
#define CC  128
#define NN  2304      // H*W = 48*48
#define GRP 32

typedef __attribute__((ext_vector_type(16))) _Float16 v16h;
typedef __attribute__((ext_vector_type(8)))  _Float16 v8h;
typedef __attribute__((ext_vector_type(8)))  float    v8f;
typedef __attribute__((ext_vector_type(4)))  int      v4i;

#if __has_builtin(__builtin_amdgcn_global_load_async_to_lds_b128)
#define HAVE_ASYNC_LDS 1
#else
#define HAVE_ASYNC_LDS 0
#endif

// Stage one 16-byte chunk global -> LDS (async on gfx1250 if available).
static __device__ __forceinline__ void stage16(const _Float16* g, _Float16* l) {
#if HAVE_ASYNC_LDS
  __builtin_amdgcn_global_load_async_to_lds_b128(
      (__attribute__((address_space(1))) v4i*)(_Float16*)g,
      (__attribute__((address_space(3))) v4i*)l, 0, 0);
#else
  *(v8h*)l = *(const v8h*)g;
#endif
}

static __device__ __forceinline__ void async_wait() {
#if HAVE_ASYNC_LDS
#if __has_builtin(__builtin_amdgcn_s_wait_asynccnt)
  __builtin_amdgcn_s_wait_asynccnt(0);
#else
  asm volatile("s_wait_asynccnt 0x0" ::: "memory");
#endif
#endif
}

static __device__ __forceinline__ v8f wmma_f16(v16h a, v16h b, v8f c) {
  // D = A(16x32) * B(32x16) + C, f32 accumulate
  return __builtin_amdgcn_wmma_f32_16x16x32_f16(false, a, false, b, (short)0, c, false, false);
}

// A fragment (16x32 f16) from row-major [M, lda] memory.
static __device__ __forceinline__ v16h load_a_f16(const _Float16* __restrict__ p0,
                                                  int lda, int m0, int k0, int lane) {
  int row = m0 + (lane & 15);
  int hf  = lane >> 4;
  const _Float16* p = p0 + (size_t)row * lda + k0 + hf * 8;
  v8h lo = *(const v8h*)(p);
  v8h hi = *(const v8h*)(p + 16);
  v16h r;
#pragma unroll
  for (int i = 0; i < 8; ++i) { r[i] = lo[i]; r[i + 8] = hi[i]; }
  return r;
}

// Same fragment, but from f32 row-major memory (weight matrices), cvt to f16.
static __device__ __forceinline__ v16h load_a_f32(const float* __restrict__ p0,
                                                  int lda, int m0, int k0, int lane) {
  int row = m0 + (lane & 15);
  int hf  = lane >> 4;
  const float* p = p0 + (size_t)row * lda + k0 + hf * 8;
  v16h r;
#pragma unroll
  for (int i = 0; i < 8; ++i) { r[i] = (_Float16)p[i]; r[i + 8] = (_Float16)p[16 + i]; }
  return r;
}

// B fragment (32x16 f16) from "N-major" memory mem[n][k] with leading dim ldb.
// Works for both global and LDS pointers (LDS -> ds_load_b128).
static __device__ __forceinline__ v16h load_b_nmaj(const _Float16* p0,
                                                   int ldb, int k0, int n0, int lane) {
  int col = n0 + (lane & 15);
  int hf  = lane >> 4;
  const _Float16* p = p0 + (size_t)col * ldb + k0 + hf * 8;
  v8h lo = *(const v8h*)(p);
  v8h hi = *(const v8h*)(p + 16);
  v16h r;
#pragma unroll
  for (int i = 0; i < 8; ++i) { r[i] = lo[i]; r[i + 8] = hi[i]; }
  return r;
}

// ---------------- GroupNorm:  x[B,C,N] f32 -> xnT[B,N,C] f16 ----------------
__global__ __launch_bounds__(256) void gn_kernel(const float* __restrict__ x,
                                                 const float* __restrict__ gw,
                                                 const float* __restrict__ gb,
                                                 _Float16* __restrict__ xnT) {
  int b = blockIdx.x / GRP;
  int g = blockIdx.x % GRP;
  const float* xg = x + ((size_t)b * CC + g * 4) * NN;   // 4 channels x 2304
  float sum = 0.f, sq = 0.f;
  for (int i = threadIdx.x; i < 4 * NN; i += 256) {
    float v = xg[i]; sum += v; sq += v * v;
  }
  __shared__ float s1[8], s2[8];
  int lane = threadIdx.x & 31, wid = threadIdx.x >> 5;
#pragma unroll
  for (int o = 16; o > 0; o >>= 1) { sum += __shfl_down(sum, o, 32); sq += __shfl_down(sq, o, 32); }
  if (lane == 0) { s1[wid] = sum; s2[wid] = sq; }
  __syncthreads();
  if (wid == 0) {
    float a = lane < 8 ? s1[lane] : 0.f;
    float c = lane < 8 ? s2[lane] : 0.f;
#pragma unroll
    for (int o = 4; o > 0; o >>= 1) { a += __shfl_down(a, o, 32); c += __shfl_down(c, o, 32); }
    if (lane == 0) {
      float mu = a / (float)(4 * NN);
      float var = c / (float)(4 * NN) - mu * mu;
      s1[0] = mu; s2[0] = rsqrtf(var + 1e-5f);
    }
  }
  __syncthreads();
  float mu = s1[0], rs = s2[0];
  for (int i = threadIdx.x; i < 4 * NN; i += 256) {
    int cl = i / NN, n = i % NN;
    int ch = g * 4 + cl;
    float y = (xg[i] - mu) * rs * gw[ch] + gb[ch];
    xnT[((size_t)b * NN + n) * CC + ch] = (_Float16)y;
  }
}

// ------------- QKV projections: [128x128] @ xnT -> qT,kT [N,C] ; v [C,N] -------------
__global__ __launch_bounds__(256) void qkv_kernel(const _Float16* __restrict__ xnT,
                                                  const float* __restrict__ wq, const float* __restrict__ bq,
                                                  const float* __restrict__ wk, const float* __restrict__ bk,
                                                  const float* __restrict__ wv, const float* __restrict__ bv,
                                                  _Float16* __restrict__ qT, _Float16* __restrict__ kT,
                                                  _Float16* __restrict__ vC) {
  int w = (blockIdx.x * blockDim.x + threadIdx.x) >> 5;   // global wave id
  int lane = threadIdx.x & 31;
  int nt = w % 36; w /= 36;        // 2304/64
  int mt = w % 8;  w /= 8;         // 128/16
  int b  = w % 8;  w /= 8;
  int proj = w;                    // 0=q 1=k 2=v
  const float* W    = proj == 0 ? wq : (proj == 1 ? wk : wv);
  const float* bias = proj == 0 ? bq : (proj == 1 ? bk : bv);
  const _Float16* xb = xnT + (size_t)b * NN * CC;
  int m0 = mt * 16, n0 = nt * 64;
  v8f acc[4];
#pragma unroll
  for (int s = 0; s < 4; ++s) acc[s] = (v8f){0.f,0.f,0.f,0.f,0.f,0.f,0.f,0.f};
#pragma unroll
  for (int k0 = 0; k0 < CC; k0 += 32) {
    v16h a = load_a_f32(W, CC, m0, k0, lane);
#pragma unroll
    for (int s = 0; s < 4; ++s) {
      v16h bf = load_b_nmaj(xb, CC, k0, n0 + s * 16, lane);
      acc[s] = wmma_f16(a, bf, acc[s]);
    }
  }
  int hf = lane >> 4, ln = lane & 15;
  float scale = (proj == 0) ? 0.08838834764831845f : 1.0f;   // fold 1/sqrt(C) into q
  if (proj < 2) {
    _Float16* outp = (proj == 0 ? qT : kT) + (size_t)b * NN * CC;
#pragma unroll
    for (int s = 0; s < 4; ++s) {
      int n = n0 + s * 16 + ln;
      v8h pack;
#pragma unroll
      for (int i = 0; i < 8; ++i)
        pack[i] = (_Float16)((acc[s][i] + bias[m0 + 8 * hf + i]) * scale);
      *(v8h*)(outp + (size_t)n * CC + m0 + 8 * hf) = pack;   // [N,C]: 16B store
    }
  } else {
    _Float16* outp = vC + (size_t)b * CC * NN;               // [C,N]
#pragma unroll
    for (int s = 0; s < 4; ++s) {
      int n = n0 + s * 16 + ln;
#pragma unroll
      for (int i = 0; i < 8; ++i) {
        int m = m0 + 8 * hf + i;
        outp[(size_t)m * NN + n] = (_Float16)(acc[s][i] + bias[m]);
      }
    }
  }
}

// ------------- scores: S[hwq,hwk] = qT @ kT^T  (K = C = 128) -> f16 [B,N,N] -------------
// Block: 8 waves, 128x128 output tile. Whole K-panel of kT (128 rows x 128 k, 32KB)
// staged once in LDS (async), padded rows (136 f16) to avoid bank conflicts.
#define SLDB 136
__global__ __launch_bounds__(256) void scores_kernel(const _Float16* __restrict__ qT,
                                                     const _Float16* __restrict__ kT,
                                                     _Float16* __restrict__ sc) {
  __shared__ _Float16 bpan[128 * SLDB];           // ~34 KB
  int wid = threadIdx.x >> 5, lane = threadIdx.x & 31;
  int blk = blockIdx.x;
  int ntb = blk % 18; blk /= 18;                  // 2304/128
  int mtb = blk % 18; blk /= 18;
  int b   = blk;
  const _Float16* qb = qT + (size_t)b * NN * CC;
  const _Float16* kb = kT + (size_t)b * NN * CC;
  int m0 = mtb * 128 + wid * 16;
  int n0 = ntb * 128;

  // stage B panel: rows n0..n0+127 of kT (each 128 f16) into padded LDS rows
  const _Float16* src = kb + (size_t)n0 * CC;     // contiguous 32KB span
  for (int j = threadIdx.x; j < 2048; j += 256) { // 16B chunks
    int r = j >> 4, kk = (j & 15) * 8;
    stage16(src + (size_t)r * CC + kk, bpan + r * SLDB + kk);
  }
  async_wait();
  __syncthreads();

  v8f acc[8];
#pragma unroll
  for (int s = 0; s < 8; ++s) acc[s] = (v8f){0.f,0.f,0.f,0.f,0.f,0.f,0.f,0.f};
#pragma unroll
  for (int k0 = 0; k0 < CC; k0 += 32) {
    v16h a = load_a_f16(qb, CC, m0, k0, lane);
#pragma unroll
    for (int s = 0; s < 8; ++s) {
      v16h bf = load_b_nmaj(bpan, SLDB, k0, s * 16, lane);  // ds_load_b128 x2
      acc[s] = wmma_f16(a, bf, acc[s]);
    }
  }
  _Float16* sb = sc + (size_t)b * NN * NN;
  int hf = lane >> 4, ln = lane & 15;
#pragma unroll
  for (int s = 0; s < 8; ++s) {
    int n = n0 + s * 16 + ln;
#pragma unroll
    for (int i = 0; i < 8; ++i)
      sb[(size_t)(m0 + 8 * hf + i) * NN + n] = (_Float16)acc[s][i];
  }
}

// ------------- softmax over query-h axis (48 elems, stride 48*N) -------------
__global__ __launch_bounds__(256) void softmax_kernel(_Float16* __restrict__ sc) {
  int idx = blockIdx.x * 256 + threadIdx.x;       // B*48*2304 total
  int nk = idx % NN;  int t = idx / NN;
  int wq = t % 48;    int b = t / 48;
  _Float16* col = sc + (size_t)b * NN * NN + (size_t)wq * NN + nk;
  const size_t stride = (size_t)48 * NN;          // step in h
  float v[48];
  float mx = -3.0e38f;
#pragma unroll
  for (int h = 0; h < 48; ++h) { v[h] = (float)col[h * stride]; mx = fmaxf(mx, v[h]); }
  float sum = 0.f;
#pragma unroll
  for (int h = 0; h < 48; ++h) { float e = __expf(v[h] - mx); v[h] = e; sum += e; }
  float inv = 1.f / sum;
#pragma unroll
  for (int h = 0; h < 48; ++h) col[h * stride] = (_Float16)(v[h] * inv);
}

// ------------- ctx: ctxT[hwq,c] = attn[hwq,:] @ v[c,:]^T  (K = N = 2304) -------------
// Block: 8 waves, 128 hwq x 128 c tile. V K-panels (128 c x 32 k, 8KB) double-buffered
// in LDS via async loads; padded rows (40 f16) against bank conflicts.
#define CLDB 40
__global__ __launch_bounds__(256) void ctx_kernel(const _Float16* __restrict__ attn,
                                                  const _Float16* __restrict__ vC,
                                                  _Float16* __restrict__ ctxT) {
  __shared__ _Float16 bpan[2][128 * CLDB];        // 2 x 10 KB
  int wid = threadIdx.x >> 5, lane = threadIdx.x & 31;
  int mtb = blockIdx.x % 18;                      // 2304/128
  int b   = blockIdx.x / 18;
  const _Float16* ab = attn + (size_t)b * NN * NN;
  const _Float16* vb = vC   + (size_t)b * CC * NN;
  int m0 = mtb * 128 + wid * 16;

  // stage step 0: 512 16B chunks (4 per c-row)
#pragma unroll
  for (int r = 0; r < 2; ++r) {
    int chunk = threadIdx.x + r * 256;
    int c = chunk >> 2, kk = (chunk & 3) * 8;
    stage16(vb + (size_t)c * NN + kk, bpan[0] + c * CLDB + kk);
  }

  v8f acc[8];
#pragma unroll
  for (int s = 0; s < 8; ++s) acc[s] = (v8f){0.f,0.f,0.f,0.f,0.f,0.f,0.f,0.f};

  for (int step = 0; step < 72; ++step) {
    int k0 = step * 32;
    async_wait();
    __syncthreads();                              // current buffer ready
    if (step + 1 < 72) {                          // prefetch next K-panel
      _Float16* nb = bpan[(step + 1) & 1];
#pragma unroll
      for (int r = 0; r < 2; ++r) {
        int chunk = threadIdx.x + r * 256;
        int c = chunk >> 2, kk = (chunk & 3) * 8;
        stage16(vb + (size_t)c * NN + k0 + 32 + kk, nb + c * CLDB + kk);
      }
    }
    const _Float16* cur = bpan[step & 1];
    v16h a = load_a_f16(ab, NN, m0, k0, lane);
#pragma unroll
    for (int s = 0; s < 8; ++s) {
      v16h bf = load_b_nmaj(cur, CLDB, 0, s * 16, lane);
      acc[s] = wmma_f16(a, bf, acc[s]);
    }
    __syncthreads();                              // all reads done before overwrite
  }

  _Float16* ob = ctxT + (size_t)b * NN * CC;
  int hf = lane >> 4, ln = lane & 15;
#pragma unroll
  for (int s = 0; s < 8; ++s) {
    int c = s * 16 + ln;
#pragma unroll
    for (int i = 0; i < 8; ++i)
      ob[(size_t)(m0 + 8 * hf + i) * CC + c] = (_Float16)acc[s][i];
  }
}

// ------------- out = x + wo @ ctx + bo  (f32 output) -------------
__global__ __launch_bounds__(256) void out_kernel(const _Float16* __restrict__ ctxT,
                                                  const float* __restrict__ wo,
                                                  const float* __restrict__ bo,
                                                  const float* __restrict__ x,
                                                  float* __restrict__ out) {
  int w = (blockIdx.x * blockDim.x + threadIdx.x) >> 5;
  int lane = threadIdx.x & 31;
  int nt = w % 36; w /= 36;         // 2304/64
  int mt = w % 8;  w /= 8;          // 128/16
  int b  = w;
  const _Float16* cb = ctxT + (size_t)b * NN * CC;
  int m0 = mt * 16, n0 = nt * 64;
  v8f acc[4];
#pragma unroll
  for (int s = 0; s < 4; ++s) acc[s] = (v8f){0.f,0.f,0.f,0.f,0.f,0.f,0.f,0.f};
#pragma unroll
  for (int k0 = 0; k0 < CC; k0 += 32) {
    v16h a = load_a_f32(wo, CC, m0, k0, lane);
#pragma unroll
    for (int s = 0; s < 4; ++s) {
      v16h bf = load_b_nmaj(cb, CC, k0, n0 + s * 16, lane);
      acc[s] = wmma_f16(a, bf, acc[s]);
    }
  }
  int hf = lane >> 4, ln = lane & 15;
#pragma unroll
  for (int s = 0; s < 4; ++s) {
    int n = n0 + s * 16 + ln;
#pragma unroll
    for (int i = 0; i < 8; ++i) {
      int m = m0 + 8 * hf + i;
      size_t idx = ((size_t)b * CC + m) * NN + n;
      out[idx] = x[idx] + acc[s][i] + bo[m];
    }
  }
}

extern "C" void kernel_launch(void* const* d_in, const int* in_sizes, int n_in,
                              void* d_out, int out_size, void* d_ws, size_t ws_size,
                              hipStream_t stream) {
  const float* x   = (const float*)d_in[0];
  const float* gnw = (const float*)d_in[1];
  const float* gnb = (const float*)d_in[2];
  const float* wq  = (const float*)d_in[3];
  const float* bq  = (const float*)d_in[4];
  const float* wk  = (const float*)d_in[5];
  const float* bk  = (const float*)d_in[6];
  const float* wv  = (const float*)d_in[7];
  const float* bv  = (const float*)d_in[8];
  const float* wo  = (const float*)d_in[9];
  const float* bo  = (const float*)d_in[10];
  float* out = (float*)d_out;

  const size_t EL = (size_t)BB * NN * CC;    // 2,359,296 elements per activation buffer
  _Float16* xnT  = (_Float16*)d_ws;
  _Float16* qT   = xnT  + EL;
  _Float16* kT   = qT   + EL;
  _Float16* vC   = kT   + EL;
  _Float16* ctxT = vC   + EL;
  _Float16* attn = ctxT + EL;                // B*N*N f16 (85 MB)

  gn_kernel     <<<BB * GRP, 256, 0, stream>>>(x, gnw, gnb, xnT);
  qkv_kernel    <<<864,  256, 0, stream>>>(xnT, wq, bq, wk, bk, wv, bv, qT, kT, vC);
  scores_kernel <<<2592, 256, 0, stream>>>(qT, kT, attn);     // 8 * 18 * 18
  softmax_kernel<<<3456, 256, 0, stream>>>(attn);
  ctx_kernel    <<<144,  256, 0, stream>>>(attn, vC, ctxT);   // 8 * 18
  out_kernel    <<<288,  256, 0, stream>>>(ctxT, wo, bo, x, out);
}
